// pyFLASH_24026047054602
// MI455X (gfx1250) — matrix operand order repaired and checked by reading, not compiled, not run on hardware
//
#include <hip/hip_runtime.h>
#include <hip/hip_bf16.h>
#include <math.h>

#define DIMD 512
#define DKD 128
#define GSZ 256
#define HIDD 2048
#define NSEQ 4096
#define NBATCH 8
#define NTOK (NBATCH * NSEQ)

typedef __bf16 bf16_t;
typedef bf16_t v16bf __attribute__((ext_vector_type(16)));
typedef float v8f __attribute__((ext_vector_type(8)));
typedef unsigned short u16;
typedef u16 ushort8 __attribute__((ext_vector_type(8)));

union FragBF { v16bf v; ushort8 u[2]; };

__device__ __forceinline__ u16 f2bf(float f) {
    unsigned u = __float_as_uint(f);
    u += 0x7fffu + ((u >> 16) & 1u);            // round-to-nearest-even
    return (u16)(u >> 16);
}
__device__ __forceinline__ float bf2f(u16 h) {
    return __uint_as_float(((unsigned)h) << 16);
}
__device__ __forceinline__ float siluf(float x) {
    return x * (1.0f / (1.0f + expf(-x)));
}

__device__ __forceinline__ v8f wmma_bf16(v16bf a, v16bf b, v8f c) {
    // v_wmma_f32_16x16x32_bf16  D = A(16x32) * B(32x16) + C
    return __builtin_amdgcn_wmma_f32_16x16x32_bf16(
        false, a, false, b, (short)0, c, false, false);
}

// A fragment (16x32 bf16, row-major source).
// ISA 7.12.2: lane<16 -> row m, K = k0+{0..7} (vgpr0-3) and k0+16+{0..7}
// (vgpr4-7); lanes 16-31 same rows, K offset +8.
__device__ __forceinline__ v16bf fragA(const u16* base, int row0, int stride,
                                       int k0, int idx, int hl) {
    const u16* p = base + (size_t)(row0 + idx) * stride + k0 + hl * 8;
    FragBF f;
    f.u[0] = *(const ushort8*)p;        // 16B aligned
    f.u[1] = *(const ushort8*)(p + 16); // 16B aligned
    return f.v;
}
// B fragment (32x16 bf16) from B^T row-major: lane = column n, elements
// K = k0+{0..7} and k0+16+{0..7}; lanes 16-31 K offset +8 (same as A).
__device__ __forceinline__ v16bf fragB(const u16* base, int row0, int stride,
                                       int k0, int idx, int hl) {
    const u16* p = base + (size_t)(row0 + idx) * stride + k0 + hl * 8;
    FragBF f;
    f.u[0] = *(const ushort8*)p;
    f.u[1] = *(const ushort8*)(p + 16);
    return f.v;
}

// Async global->LDS copy of 64B per lane (4 x 16B); IOFFSET applies to both addresses
// (dsaddr = VDST + IOFFSET, gaddr = VADDR + IOFFSET), tracked by ASYNCcnt.
__device__ __forceinline__ void async_b128(unsigned lds_off, const u16* gptr) {
    asm volatile("global_load_async_to_lds_b128 %0, %1, off"
                 :: "v"(lds_off), "v"(gptr) : "memory");
    asm volatile("global_load_async_to_lds_b128 %0, %1, off offset:16"
                 :: "v"(lds_off), "v"(gptr) : "memory");
    asm volatile("global_load_async_to_lds_b128 %0, %1, off offset:32"
                 :: "v"(lds_off), "v"(gptr) : "memory");
    asm volatile("global_load_async_to_lds_b128 %0, %1, off offset:48"
                 :: "v"(lds_off), "v"(gptr) : "memory");
}
__device__ __forceinline__ unsigned lds_off32(const void* p) {
    // generic LDS address: low 32 bits are the LDS byte offset (ISA 10.2)
    return (unsigned)(size_t)p;
}

// ---------------------------------------------------------------- prep ----
__global__ void k_transpose_bf16(const float* __restrict__ in,
                                 u16* __restrict__ out, int R, int C) {
    int i = blockIdx.x * 256 + threadIdx.x;
    if (i >= R * C) return;
    int r = i / C, c = i - r * C;
    out[(size_t)c * R + r] = f2bf(in[i]);
}

__global__ void k_bias_table(const float* __restrict__ rel_bias,
                             float* __restrict__ bias_t) {
    int t = threadIdx.x;                 // one block of 512; rel = t-255
    if (t >= 511) return;
    int rel = t - 255;
    int n = -rel;
    int ret = (n < 0) ? 16 : 0;
    int na = (n < 0) ? -n : n;
    int bucket;
    if (na < 8) {
        bucket = ret + na;
    } else {
        int vl = 8 + (int)(logf((float)na * 0.125f) * (8.0f / logf(16.0f)));
        if (vl > 15) vl = 15;
        bucket = ret + vl;
    }
    bias_t[t] = rel_bias[bucket] * 16.0f;   // REL_SCALE = sqrt(256)
}

// ----------------------------------------------------------- layernorm ----
__global__ void __launch_bounds__(256)
k_layernorm(const float* __restrict__ x, const float* __restrict__ gamma,
            const float* __restrict__ beta, u16* __restrict__ xn) {
    __shared__ float s1[256], s2[256];
    int t = threadIdx.x, tok = blockIdx.x;
    const float2* px = (const float2*)(x + (size_t)tok * DIMD);
    float2 v = px[t];
    s1[t] = v.x + v.y;
    s2[t] = v.x * v.x + v.y * v.y;
    __syncthreads();
    for (int o = 128; o > 0; o >>= 1) {
        if (t < o) { s1[t] += s1[t + o]; s2[t] += s2[t + o]; }
        __syncthreads();
    }
    float mean = s1[0] * (1.0f / DIMD);
    float var  = s2[0] * (1.0f / DIMD) - mean * mean;
    float rstd = rsqrtf(var + 1e-5f);
    float2 g = ((const float2*)gamma)[t];
    float2 b = ((const float2*)beta)[t];
    u16* po = xn + (size_t)tok * DIMD + 2 * t;
    po[0] = f2bf((v.x - mean) * rstd * g.x + b.x);
    po[1] = f2bf((v.y - mean) * rstd * g.y + b.y);
}

// ------------------------------------------------- hidden GEMM + SiLU -----
// C[32768,4096] = silu(xn[32768,512] @ Wh[512,4096] + b); split v / gate.
// B^T rows staged via double-buffered async global->LDS copies.
#define HB_K 32
__global__ void __launch_bounds__(256)
k_gemm_hidden(const u16* __restrict__ xn, const u16* __restrict__ WhT,
              const float* __restrict__ b_hidden,
              u16* __restrict__ vbuf, u16* __restrict__ gate) {
    extern __shared__ u16 smem[];          // 2 x [256][HB_K] bf16 = 32 KB
    int tid = threadIdx.x;
    int wave = tid >> 5, lane = tid & 31, idx = lane & 15, hl = lane >> 4;
    int m0 = blockIdx.x * 16;
    int n0b = blockIdx.y * 256;
    u16* buf0 = smem;
    u16* buf1 = smem + 256 * HB_K;

    // prologue: stage slab 0 (thread t copies B^T row n0b+t, 64B)
    async_b128(lds_off32(buf0 + tid * HB_K),
               WhT + (size_t)(n0b + tid) * DIMD);

    v8f c0 = {0,0,0,0,0,0,0,0}, c1 = c0;
    for (int ks = 0; ks < 16; ++ks) {
        u16* cur = (ks & 1) ? buf1 : buf0;
        u16* nxt = (ks & 1) ? buf0 : buf1;
        __syncthreads();                     // prev readers of nxt are done
        if (ks + 1 < 16) {
            async_b128(lds_off32(nxt + tid * HB_K),
                       WhT + (size_t)(n0b + tid) * DIMD + (ks + 1) * HB_K);
            asm volatile("s_wait_asynccnt 4" ::: "memory");  // cur slab landed
        } else {
            asm volatile("s_wait_asynccnt 0" ::: "memory");
        }
        __syncthreads();                     // all waves' copies visible
        v16bf a  = fragA(xn, m0, DIMD, ks * HB_K, idx, hl);
        v16bf b0 = fragB(cur, wave * 32,      HB_K, 0, idx, hl);
        v16bf b1 = fragB(cur, wave * 32 + 16, HB_K, 0, idx, hl);
        c0 = wmma_bf16(a, b0, c0);
        c1 = wmma_bf16(a, b1, c1);
    }
#pragma unroll
    for (int r = 0; r < 8; ++r) {
        int row = m0 + r + 8 * hl;
        int col0 = n0b + wave * 32 + idx, col1 = col0 + 16;
        float h0 = siluf(c0[r] + b_hidden[col0]);
        float h1 = siluf(c1[r] + b_hidden[col1]);
        if (col0 < HIDD) vbuf[(size_t)row * HIDD + col0] = f2bf(h0);
        else             gate[(size_t)row * HIDD + col0 - HIDD] = f2bf(h0);
        if (col1 < HIDD) vbuf[(size_t)row * HIDD + col1] = f2bf(h1);
        else             gate[(size_t)row * HIDD + col1 - HIDD] = f2bf(h1);
    }
}

// ------------------------------------- qk GEMM + SiLU + OffsetScale -------
__global__ void __launch_bounds__(128)
k_gemm_qk(const u16* __restrict__ xn, const u16* __restrict__ WqkT,
          const float* __restrict__ b_qk, const float* __restrict__ os_gamma,
          const float* __restrict__ os_beta, const unsigned char* __restrict__ mask,
          u16* __restrict__ qq, u16* __restrict__ lq,
          u16* __restrict__ qkk, u16* __restrict__ lk) {
    int wave = threadIdx.x >> 5, lane = threadIdx.x & 31;
    int idx = lane & 15, hl = lane >> 4;
    int m0 = blockIdx.x * 16;
    int n0 = wave * 32;
    v8f c0 = {0,0,0,0,0,0,0,0}, c1 = c0;
    for (int ks = 0; ks < 16; ++ks) {
        int k0 = ks * 32;
        v16bf a  = fragA(xn,   m0,      DIMD, k0, idx, hl);
        v16bf b0 = fragB(WqkT, n0,      DIMD, k0, idx, hl);
        v16bf b1 = fragB(WqkT, n0 + 16, DIMD, k0, idx, hl);
        c0 = wmma_bf16(a, b0, c0);
        c1 = wmma_bf16(a, b1, c1);
    }
#pragma unroll
    for (int r = 0; r < 8; ++r) {
        int row = m0 + r + 8 * hl;
        float mk = (float)mask[row];
#pragma unroll
        for (int t = 0; t < 2; ++t) {
            int col = n0 + t * 16 + idx;
            float acc = (t == 0) ? c0[r] : c1[r];
            float s = siluf(acc + b_qk[col]);
            size_t o = (size_t)row * DKD + col;
            qq[o]  = f2bf(s * os_gamma[0 * DKD + col] + os_beta[0 * DKD + col]);
            lq[o]  = f2bf(s * os_gamma[1 * DKD + col] + os_beta[1 * DKD + col]);
            qkk[o] = f2bf(s * os_gamma[2 * DKD + col] + os_beta[2 * DKD + col]);
            lk[o]  = f2bf((s * os_gamma[3 * DKD + col] + os_beta[3 * DKD + col]) * mk);
        }
    }
}

// ------------------------------------------- quadratic group attention ----
// One block = half a group (128 query rows, all 256 keys). 8 waves.
#define APITCH 264   // 256 + 8 pad, 528B row -> 16B aligned fragments
__global__ void __launch_bounds__(256)
k_quad_attn(const u16* __restrict__ qq, const u16* __restrict__ qkk,
            const u16* __restrict__ vbuf, const float* __restrict__ bias_t,
            const unsigned char* __restrict__ mask, u16* __restrict__ qo) {
    extern __shared__ u16 smem[];
    u16* attn_s = smem;                 // [128][APITCH] bf16 attention tile
    u16* vgT    = smem + 128 * APITCH;  // [32][APITCH]  v^T chunk
    int wave = threadIdx.x >> 5, lane = threadIdx.x & 31;
    int idx = lane & 15, hl = lane >> 4;
    int grp = blockIdx.x >> 1, half = blockIdx.x & 1;
    int t0 = (grp >> 4) * NSEQ + (grp & 15) * GSZ;  // group token base
    int rq = t0 + half * 128;                       // query row base

    // ---- stage 1: attn = laplace(qq.qk^T/g + bias) * mask  (bf16 -> LDS)
    v16bf a[4];
#pragma unroll
    for (int ks = 0; ks < 4; ++ks)
        a[ks] = fragA(qq, rq + wave * 16, DKD, ks * 32, idx, hl);
    for (int jt = 0; jt < 16; ++jt) {
        v8f c = {0,0,0,0,0,0,0,0};
#pragma unroll
        for (int ks = 0; ks < 4; ++ks) {
            v16bf b = fragB(qkk, t0 + jt * 16, DKD, ks * 32, idx, hl);
            c = wmma_bf16(a[ks], b, c);
        }
#pragma unroll
        for (int r = 0; r < 8; ++r) {
            int il = wave * 16 + r + 8 * hl;        // local query row 0..127
            int j = jt * 16 + idx;                  // key position 0..255
            float val = c[r] * (1.0f / GSZ) + bias_t[j - (half * 128 + il) + 255];
            const float MU  = 0.70710678f;          // sqrt(0.5)
            const float INV = 0.79788456f;          // 1/(sqrt(pi/4)*sqrt(2))
            float att = 0.5f * (1.0f + erff((val - MU) * INV));
            att *= (float)mask[t0 + j];
            attn_s[il * APITCH + j] = f2bf(att);
        }
    }
    __syncthreads();

    // ---- stage 2: quad_out = attn[128,256] @ v[256,2048], 32-col chunks
    for (int ec = 0; ec < 64; ++ec) {
        int e0 = ec * 32;
        if (ec) __syncthreads();        // protect vgT reuse
        {   // stage v^T chunk: thread j copies v[t0+j, e0..e0+31] transposed
            int j = threadIdx.x;
            const u16* src = vbuf + (size_t)(t0 + j) * HIDD + e0;
#pragma unroll
            for (int s4 = 0; s4 < 4; ++s4) {
                ushort8 u = *(const ushort8*)(src + s4 * 8);
#pragma unroll
                for (int q = 0; q < 8; ++q) vgT[(s4 * 8 + q) * APITCH + j] = u[q];
            }
        }
        __syncthreads();
        v8f c0 = {0,0,0,0,0,0,0,0}, c1 = c0;
#pragma unroll
        for (int ks = 0; ks < 8; ++ks) {
            int j0 = ks * 32;
            v16bf aa = fragA(attn_s, wave * 16, APITCH, j0, idx, hl);
            v16bf b0 = fragB(vgT, 0,  APITCH, j0, idx, hl);
            v16bf b1 = fragB(vgT, 16, APITCH, j0, idx, hl);
            c0 = wmma_bf16(aa, b0, c0);
            c1 = wmma_bf16(aa, b1, c1);
        }
#pragma unroll
        for (int r = 0; r < 8; ++r) {
            int row = rq + wave * 16 + r + 8 * hl;
            qo[(size_t)row * HIDD + e0 + idx]      = f2bf(c0[r]);
            qo[(size_t)row * HIDD + e0 + 16 + idx] = f2bf(c1[r]);
        }
    }
}

// ------------------------------------------ lin_kv = lk^T @ v / n ---------
#define KPITCH 48    // 32 + 16 pad, 96B row -> 16B aligned fragments
__global__ void __launch_bounds__(256)
k_lin_kv(const u16* __restrict__ lk, const u16* __restrict__ vbuf,
         float* __restrict__ kvf) {
    extern __shared__ u16 smem[];
    u16* lkT = smem;                  // [128][KPITCH]
    u16* vT  = smem + 128 * KPITCH;   // [256][KPITCH]
    int wave = threadIdx.x >> 5, lane = threadIdx.x & 31;
    int idx = lane & 15, hl = lane >> 4;
    int batch = blockIdx.y, e0 = blockIdx.x * 256;
    int tb = batch * NSEQ;
    v8f zero = {0,0,0,0,0,0,0,0};
    v8f c[16];
#pragma unroll
    for (int et = 0; et < 16; ++et) c[et] = zero;

    for (int n0 = 0; n0 < NSEQ; n0 += 32) {
        __syncthreads();
        {   // stage lk^T: 32 rows x 128 cols
            int row = threadIdx.x >> 3, seg = threadIdx.x & 7;  // seg: 16 cols
            const u16* src = lk + (size_t)(tb + n0 + row) * DKD + seg * 16;
#pragma unroll
            for (int s4 = 0; s4 < 2; ++s4) {
                ushort8 u = *(const ushort8*)(src + s4 * 8);
#pragma unroll
                for (int q = 0; q < 8; ++q)
                    lkT[(seg * 16 + s4 * 8 + q) * KPITCH + row] = u[q];
            }
        }
        {   // stage v^T: 32 rows x 256 cols (at e0)
            int row = threadIdx.x >> 3, seg = threadIdx.x & 7;  // seg: 32 cols
            const u16* src = vbuf + (size_t)(tb + n0 + row) * HIDD + e0 + seg * 32;
#pragma unroll
            for (int s4 = 0; s4 < 4; ++s4) {
                ushort8 u = *(const ushort8*)(src + s4 * 8);
#pragma unroll
                for (int q = 0; q < 8; ++q)
                    vT[(seg * 32 + s4 * 8 + q) * KPITCH + row] = u[q];
            }
        }
        __syncthreads();
        v16bf aa = fragA(lkT, wave * 16, KPITCH, 0, idx, hl);
#pragma unroll
        for (int et = 0; et < 16; ++et) {
            v16bf bb = fragB(vT, et * 16, KPITCH, 0, idx, hl);
            c[et] = wmma_bf16(aa, bb, c[et]);
        }
    }
#pragma unroll
    for (int et = 0; et < 16; ++et) {
#pragma unroll
        for (int r = 0; r < 8; ++r) {
            int d = wave * 16 + r + 8 * hl;
            int e = e0 + et * 16 + idx;
            kvf[((size_t)batch * DKD + d) * HIDD + e] = c[et][r] * (1.0f / NSEQ);
        }
    }
}

__global__ void k_kv_transpose(const float* __restrict__ kvf,
                               u16* __restrict__ kvT) {
    int i = blockIdx.x * 256 + threadIdx.x;      // over 8*128*2048
    if (i >= NBATCH * DKD * HIDD) return;
    int b = i / (DKD * HIDD);
    int rem = i - b * DKD * HIDD;
    int d = rem / HIDD, e = rem - d * HIDD;
    kvT[((size_t)b * HIDD + e) * DKD + d] = f2bf(kvf[i]);
}

// --------------------- lin_out = lq @ lin_kv; fuse +qo and gate (in-place)
__global__ void __launch_bounds__(256)
k_lin_out_gate(const u16* __restrict__ lq, const u16* __restrict__ kvT,
               const u16* __restrict__ gate, u16* __restrict__ qo) {
    int wave = threadIdx.x >> 5, lane = threadIdx.x & 31;
    int idx = lane & 15, hl = lane >> 4;
    int m0 = blockIdx.x * 16;
    int n0 = blockIdx.y * 256 + wave * 32;
    const u16* B = kvT + (size_t)(m0 >> 12) * HIDD * DKD;   // per-batch kv^T
    v8f c0 = {0,0,0,0,0,0,0,0}, c1 = c0;
#pragma unroll
    for (int ks = 0; ks < 4; ++ks) {
        int k0 = ks * 32;
        v16bf a  = fragA(lq, m0,      DKD, k0, idx, hl);
        v16bf b0 = fragB(B,  n0,      DKD, k0, idx, hl);
        v16bf b1 = fragB(B,  n0 + 16, DKD, k0, idx, hl);
        c0 = wmma_bf16(a, b0, c0);
        c1 = wmma_bf16(a, b1, c1);
    }
#pragma unroll
    for (int r = 0; r < 8; ++r) {
        int row = m0 + r + 8 * hl;
        size_t o0 = (size_t)row * HIDD + n0 + idx;
        size_t o1 = o0 + 16;
        qo[o0] = f2bf((c0[r] + bf2f(qo[o0])) * bf2f(gate[o0]));
        qo[o1] = f2bf((c1[r] + bf2f(qo[o1])) * bf2f(gate[o1]));
    }
}

// ------------------------- out = g_out @ W_out + b_out + x (f32) ----------
// Same double-buffered async-to-LDS pipeline for W_out^T rows.
__global__ void __launch_bounds__(256)
k_out_proj(const u16* __restrict__ gout, const u16* __restrict__ WoutT,
           const float* __restrict__ b_out, const float* __restrict__ x,
           float* __restrict__ out) {
    extern __shared__ u16 smem[];          // 2 x [256][HB_K] bf16 = 32 KB
    int tid = threadIdx.x;
    int wave = tid >> 5, lane = tid & 31, idx = lane & 15, hl = lane >> 4;
    int m0 = blockIdx.x * 16;
    int n0b = blockIdx.y * 256;
    u16* buf0 = smem;
    u16* buf1 = smem + 256 * HB_K;

    async_b128(lds_off32(buf0 + tid * HB_K),
               WoutT + (size_t)(n0b + tid) * HIDD);

    v8f c0 = {0,0,0,0,0,0,0,0}, c1 = c0;
    for (int ks = 0; ks < 64; ++ks) {
        u16* cur = (ks & 1) ? buf1 : buf0;
        u16* nxt = (ks & 1) ? buf0 : buf1;
        __syncthreads();
        if (ks + 1 < 64) {
            async_b128(lds_off32(nxt + tid * HB_K),
                       WoutT + (size_t)(n0b + tid) * HIDD + (ks + 1) * HB_K);
            asm volatile("s_wait_asynccnt 4" ::: "memory");
        } else {
            asm volatile("s_wait_asynccnt 0" ::: "memory");
        }
        __syncthreads();
        __builtin_prefetch(gout + (size_t)(m0 + idx) * HIDD + ks * HB_K + 256, 0, 0);
        v16bf a  = fragA(gout, m0, HIDD, ks * HB_K, idx, hl);
        v16bf b0 = fragB(cur, wave * 32,      HB_K, 0, idx, hl);
        v16bf b1 = fragB(cur, wave * 32 + 16, HB_K, 0, idx, hl);
        c0 = wmma_bf16(a, b0, c0);
        c1 = wmma_bf16(a, b1, c1);
    }
#pragma unroll
    for (int r = 0; r < 8; ++r) {
        int row = m0 + r + 8 * hl;
        int col0 = n0b + wave * 32 + idx, col1 = col0 + 16;
        size_t o0 = (size_t)row * DIMD + col0;
        size_t o1 = (size_t)row * DIMD + col1;
        out[o0] = c0[r] + b_out[col0] + x[o0];
        out[o1] = c1[r] + b_out[col1] + x[o1];
    }
}

// ==========================================================================
extern "C" void kernel_launch(void* const* d_in, const int* in_sizes, int n_in,
                              void* d_out, int out_size, void* d_ws, size_t ws_size,
                              hipStream_t stream) {
    const float* x        = (const float*)d_in[0];
    const unsigned char* mask = (const unsigned char*)d_in[1];
    const float* ln_gamma = (const float*)d_in[2];
    const float* ln_beta  = (const float*)d_in[3];
    const float* W_hidden = (const float*)d_in[4];
    const float* b_hidden = (const float*)d_in[5];
    const float* W_qk     = (const float*)d_in[6];
    const float* b_qk     = (const float*)d_in[7];
    const float* os_gamma = (const float*)d_in[8];
    const float* os_beta  = (const float*)d_in[9];
    const float* rel_bias = (const float*)d_in[10];
    const float* W_out    = (const float*)d_in[11];
    const float* b_out    = (const float*)d_in[12];
    float* out = (float*)d_out;

    char* ws = (char*)d_ws;
    size_t off = 0;
    auto alloc = [&](size_t bytes) -> char* {
        char* p = ws + off;
        off += (bytes + 255) & ~(size_t)255;
        return p;
    };
    u16* WhT    = (u16*)alloc((size_t)(2 * HIDD) * DIMD * 2); // [4096][512]
    u16* WqkT   = (u16*)alloc((size_t)DKD * DIMD * 2);        // [128][512]
    u16* WoutT  = (u16*)alloc((size_t)DIMD * HIDD * 2);       // [512][2048]
    float* biasT = (float*)alloc(512 * 4);
    u16* xn     = (u16*)alloc((size_t)NTOK * DIMD * 2);
    u16* vbuf   = (u16*)alloc((size_t)NTOK * HIDD * 2);
    u16* gate   = (u16*)alloc((size_t)NTOK * HIDD * 2);
    u16* qq     = (u16*)alloc((size_t)NTOK * DKD * 2);
    u16* lq     = (u16*)alloc((size_t)NTOK * DKD * 2);
    u16* qkk    = (u16*)alloc((size_t)NTOK * DKD * 2);
    u16* lk     = (u16*)alloc((size_t)NTOK * DKD * 2);
    u16* qo     = (u16*)alloc((size_t)NTOK * HIDD * 2);  // also holds g_out
    float* kvf  = (float*)alloc((size_t)NBATCH * DKD * HIDD * 4);
    u16* kvT    = (u16*)alloc((size_t)NBATCH * HIDD * DKD * 2);
    (void)ws_size; (void)in_sizes; (void)n_in; (void)out_size;

    // weight prep
    k_transpose_bf16<<<(DIMD * 2 * HIDD + 255) / 256, 256, 0, stream>>>(
        W_hidden, WhT, DIMD, 2 * HIDD);
    k_transpose_bf16<<<(DIMD * DKD + 255) / 256, 256, 0, stream>>>(
        W_qk, WqkT, DIMD, DKD);
    k_transpose_bf16<<<(HIDD * DIMD + 255) / 256, 256, 0, stream>>>(
        W_out, WoutT, HIDD, DIMD);
    k_bias_table<<<1, 512, 0, stream>>>(rel_bias, biasT);

    // layernorm -> bf16
    k_layernorm<<<NTOK, 256, 0, stream>>>(x, ln_gamma, ln_beta, xn);

    // hidden + qk projections
    size_t smemH = (size_t)2 * 256 * HB_K * sizeof(u16);   // 32 KB
    k_gemm_hidden<<<dim3(NTOK / 16, (2 * HIDD) / 256), 256, smemH, stream>>>(
        xn, WhT, b_hidden, vbuf, gate);
    k_gemm_qk<<<dim3(NTOK / 16), 128, 0, stream>>>(
        xn, WqkT, b_qk, os_gamma, os_beta, mask, qq, lq, qkk, lk);

    // quadratic attention: 2 blocks per (batch,group) half
    size_t smem4 = (size_t)(128 * APITCH + 32 * APITCH) * sizeof(u16); // ~84.5KB
    k_quad_attn<<<NBATCH * (NSEQ / GSZ) * 2, 256, smem4, stream>>>(
        qq, qkk, vbuf, biasT, mask, qo);

    // linear attention kv + transpose
    size_t smem5 = (size_t)(128 * KPITCH + 256 * KPITCH) * sizeof(u16); // 36KB
    k_lin_kv<<<dim3(HIDD / 256, NBATCH), 256, smem5, stream>>>(lk, vbuf, kvf);
    k_kv_transpose<<<(NBATCH * DKD * HIDD + 255) / 256, 256, 0, stream>>>(kvf, kvT);

    // lin_out + qo, gated (in-place on qo)
    k_lin_out_gate<<<dim3(NTOK / 16, HIDD / 256), 256, 0, stream>>>(
        lq, kvT, gate, qo);

    // output projection + bias + residual
    k_out_proj<<<dim3(NTOK / 16, DIMD / 256), 256, smemH, stream>>>(
        qo, WoutT, b_out, x, out);
}
